// unif_vMF_12128987644656
// MI455X (gfx1250) — matrix operand order repaired and checked by reading, not compiled, not run on hardware
//
#include <hip/hip_runtime.h>

// ---------------------------------------------------------------------------
// vMF sampler for MI455X (gfx1250, wave32, WMMA).
//   Stage 1: mu_raw = lat_code @ W.T + b   via v_wmma_f32_16x16x32_bf16,
//            double-buffered LDS pipeline, v_cvt_pk_bf16_f32 conversion.
//   Stage 2: row-wise normalize + vMF sampling epilogue (memory bound)
// GEMM: M=65536, N=512, K=1024. bf16 inputs / f32 accumulate.
// ---------------------------------------------------------------------------

#define HID   1024
#define LATD  512
#define BATCH 65536

#define TILE_M 128
#define TILE_N 128
#define TILE_K 32
#define LDSS   40      // padded LDS row stride in bf16 elements (80 bytes)

typedef __attribute__((ext_vector_type(16))) __bf16 v16bf;
typedef __attribute__((ext_vector_type(8)))  float  v8f;

union FragB16 { v16bf v; uint4 q[2]; };

// f32 pair -> packed bf16 dword (RNE) in one instruction
__device__ __forceinline__ unsigned int pack_bf16(float a, float b) {
    unsigned int r;
    asm volatile("v_cvt_pk_bf16_f32 %0, %1, %2" : "=v"(r) : "v"(a), "v"(b));
    return r;
}

// ---------------------------------------------------------------------------
// GEMM + bias: out[m,n] = sum_k A[m,k]*W[n,k] + bias[n]
// grid = (N/128, M/128), block = 256 (8 waves: 4 along M x 2 along N)
// ---------------------------------------------------------------------------
__device__ __forceinline__ void load_tile_regs(
    const float* __restrict__ A, const float* __restrict__ W,
    int m_base, int n_base, int k0, int rr0, int kq,
    float4 va[4], float4 vb[4])
{
#pragma unroll
    for (int i = 0; i < 4; ++i) {
        const int r = rr0 + i * 32;
        va[i] = *(const float4*)(A + (size_t)(m_base + r) * HID + k0 + kq * 4);
        vb[i] = *(const float4*)(W + (size_t)(n_base + r) * HID + k0 + kq * 4);
    }
}

__device__ __forceinline__ void store_tile_lds(
    unsigned short* smA, unsigned short* smB,
    int rr0, int kq, const float4 va[4], const float4 vb[4])
{
#pragma unroll
    for (int i = 0; i < 4; ++i) {
        const int r = rr0 + i * 32;
        uint2 pa; pa.x = pack_bf16(va[i].x, va[i].y); pa.y = pack_bf16(va[i].z, va[i].w);
        uint2 pb; pb.x = pack_bf16(vb[i].x, vb[i].y); pb.y = pack_bf16(vb[i].z, vb[i].w);
        *(uint2*)&smA[r * LDSS + kq * 4] = pa;
        *(uint2*)&smB[r * LDSS + kq * 4] = pb;
    }
}

__global__ __launch_bounds__(256) void gemm_bias_wmma(
    const float* __restrict__ A,     // [BATCH, HID]
    const float* __restrict__ W,     // [LATD, HID]
    const float* __restrict__ bias,  // [LATD]
    float* __restrict__ out)         // [BATCH, LATD]  (mu_raw)
{
    __shared__ __align__(16) unsigned short smA[2][TILE_M * LDSS];
    __shared__ __align__(16) unsigned short smB[2][TILE_N * LDSS];

    const int tid   = threadIdx.x;
    const int lane  = tid & 31;
    const int wave  = tid >> 5;
    const int waveM = wave & 3;    // 0..3
    const int waveN = wave >> 2;   // 0..1
    const int half  = lane >> 4;   // 0..1
    const int lrow  = lane & 15;   // 0..15

    const int m_base = blockIdx.y * TILE_M;
    const int n_base = blockIdx.x * TILE_N;

    v8f acc[2][4];
#pragma unroll
    for (int i = 0; i < 2; ++i)
#pragma unroll
        for (int j = 0; j < 4; ++j)
#pragma unroll
            for (int r = 0; r < 8; ++r) acc[i][j][r] = 0.0f;

    const int kq  = tid & 7;   // which float4 chunk within the 32-wide K slab
    const int rr0 = tid >> 3;  // 0..31 base row

    // pipeline prologue: stage K-slab 0 into LDS buffer 0
    float4 va[4], vb[4];
    load_tile_regs(A, W, m_base, n_base, 0, rr0, kq, va, vb);
    store_tile_lds(smA[0], smB[0], rr0, kq, va, vb);
    __syncthreads();

    for (int k0 = 0, buf = 0; k0 < HID; k0 += TILE_K, buf ^= 1) {
        const bool more = (k0 + TILE_K) < HID;
        if (more) {
            // issue next slab's global loads before the WMMAs (latency overlap)
            load_tile_regs(A, W, m_base, n_base, k0 + TILE_K, rr0, kq, va, vb);
            if (k0 + 2 * TILE_K < HID) {
                // warm L2 for the slab after next (A streams once from HBM)
                __builtin_prefetch((const void*)(A + (size_t)(m_base + rr0) * HID + k0 + 2 * TILE_K + kq * 4), 0, 0);
            }
        }

        // fragment loads per ISA VGPR layouts (wave32, 16-bit A 16x32 / B 32x16)
        const unsigned short* sA = smA[buf];
        const unsigned short* sB = smB[buf];
        FragB16 af[2], bfr[4];
#pragma unroll
        for (int mf = 0; mf < 2; ++mf) {
            const int row = waveM * 32 + mf * 16 + lrow;
            af[mf].q[0] = *(const uint4*)&sA[row * LDSS + half * 8];        // K = half*8 + 0..7
            af[mf].q[1] = *(const uint4*)&sA[row * LDSS + 16 + half * 8];   // K = 16 + half*8 + 0..7
        }
#pragma unroll
        for (int nf = 0; nf < 4; ++nf) {
            const int row = waveN * 64 + nf * 16 + lrow;
            bfr[nf].q[0] = *(const uint4*)&sB[row * LDSS + half * 8];       // K = half*8 + 0..7
            bfr[nf].q[1] = *(const uint4*)&sB[row * LDSS + 16 + half * 8];  // K = 16 + half*8 + 0..7
        }

#pragma unroll
        for (int mf = 0; mf < 2; ++mf)
#pragma unroll
            for (int nf = 0; nf < 4; ++nf)
                acc[mf][nf] = __builtin_amdgcn_wmma_f32_16x16x32_bf16(
                    false, af[mf].v, false, bfr[nf].v,
                    (short)0, acc[mf][nf], false, false);

        if (more) {
            // convert + store into the other buffer while WMMAs drain
            store_tile_lds(smA[buf ^ 1], smB[buf ^ 1], rr0, kq, va, vb);
        }
        __syncthreads();
    }

    // bias + store (D layout: VGPR r -> M = half*8 + r, N = lrow within 16x16 tile)
#pragma unroll
    for (int nf = 0; nf < 4; ++nf) {
        const int n  = n_base + waveN * 64 + nf * 16 + lrow;
        const float bv = bias[n];
#pragma unroll
        for (int mf = 0; mf < 2; ++mf) {
            const int mrow = m_base + waveM * 32 + mf * 16 + half * 8;
#pragma unroll
            for (int r = 0; r < 8; ++r)
                out[(size_t)(mrow + r) * LATD + n] = acc[mf][nf][r] + bv;
        }
    }
}

// ---------------------------------------------------------------------------
// Row-wise epilogue: normalize, vMF sample, outputs.
// ---------------------------------------------------------------------------
__device__ __forceinline__ unsigned int ihash(unsigned int a) {
    a ^= a >> 16; a *= 0x7FEB352Du;
    a ^= a >> 15; a *= 0x846CA68Bu;
    a ^= a >> 16; return a;
}
__device__ __forceinline__ float u01f(unsigned int h) {
    return ((h >> 8) + 0.5f) * (1.0f / 16777216.0f);
}
__device__ __forceinline__ float gaussf(unsigned int s) {
    const float u1 = u01f(ihash(s));
    const float u2 = u01f(ihash(s ^ 0x68E31DA4u));
    return sqrtf(-2.0f * __logf(u1)) * __cosf(6.28318530718f * u2);
}
__device__ __forceinline__ float blockSum(float v, volatile float* sm) {
#pragma unroll
    for (int o = 16; o > 0; o >>= 1) v += __shfl_xor(v, o, 32);
    __syncthreads();                       // guard previous use of sm
    if ((threadIdx.x & 31) == 0) sm[threadIdx.x >> 5] = v;
    __syncthreads();
    float t = 0.0f;
#pragma unroll
    for (int i = 0; i < 8; ++i) t += sm[i];
    return t;
}

// vmf_kld(kappa=1, d=512) + log(10/1)  (host-side constant in the reference)
#define KLD_CONST 2.3035517f

__global__ __launch_bounds__(256) void vmf_epilogue(
    const float* __restrict__ muraw,    // [BATCH, LATD] (aliases out_mu; read then overwrite)
    float* __restrict__ out_sampled,
    float* __restrict__ out_mu,
    float* __restrict__ out_norm,
    float* __restrict__ out_kld)
{
    __shared__ float smred[8];
    __shared__ float smrow[2];           // [0]=w, [1]=munoise
    const int row = blockIdx.x;
    const int t   = threadIdx.x;
    const size_t base = (size_t)row * LATD;

    const float x0 = muraw[base + t];
    const float x1 = muraw[base + t + 256];

    const float norm = sqrtf(blockSum(x0 * x0 + x1 * x1, smred));
    const float inv  = 1.0f / norm;
    const float m0 = x0 * inv, m1 = x1 * inv;   // mu (unit; munorm==1 after this)

    // per-element tangent noise v ~ N(0,1), deterministic hash RNG
    const float v0 = gaussf(ihash((unsigned)row * 0x9E3779B9u) ^ (unsigned)t);
    const float v1 = gaussf(ihash((unsigned)row * 0x9E3779B9u) ^ (unsigned)(t + 256));

    const float proj = blockSum(m0 * v0 + m1 * v1, smred);
    float o0 = v0 - m0 * proj;
    float o1 = v1 - m1 * proj;
    const float oinv = 1.0f / sqrtf(blockSum(o0 * o0 + o1 * o1, smred));
    o0 *= oinv; o1 *= oinv;

    if (t == 0) {
        // Wood (1994) rejection sampler, kappa=1, dim=511.
        // Beta(255.5,255.5) ~= N(0.5, 1/sqrt(4*512)) for large dim.
        const float kappa = 1.0f;
        const float dim   = 511.0f;
        const float bb = dim / (sqrtf(4.0f + dim * dim) + 2.0f);
        const float xx = (1.0f - bb) / (1.0f + bb);
        const float cc = kappa * xx + dim * __logf(1.0f - xx * xx);
        float w = 0.0f;
        unsigned int s = ihash(0xDEADBEEFu ^ (unsigned)row);
        for (int it = 0; it < 16; ++it) {
            s = ihash(s + 0x632BE5ABu);
            float z = 0.5f + 0.0220970869f * gaussf(s);
            z = fminf(fmaxf(z, 1e-6f), 1.0f - 1e-6f);
            const float wn = (1.0f - (1.0f + bb) * z) / (1.0f - (1.0f - bb) * z);
            s = ihash(s + 0x85EBCA6Bu);
            const float u = u01f(s);
            if (kappa * wn + dim * __logf(1.0f - xx * wn) - cc >= __logf(u)) { w = wn; break; }
        }
        smrow[0] = w;
        smrow[1] = 1.0f + u01f(ihash((unsigned)row ^ 0xA511E9B3u));  // clip(1,0,9) + U*eps
        out_norm[row] = norm;
        out_kld[row]  = KLD_CONST;
    }
    __syncthreads();
    const float w = smrow[0], munoise = smrow[1];
    const float sw = sqrtf(fmaxf(0.0f, 1.0f - w * w));

    out_sampled[base + t]       = (o0 * sw + m0 * w) * munoise;
    out_sampled[base + t + 256] = (o1 * sw + m1 * w) * munoise;
    out_mu[base + t]       = m0;
    out_mu[base + t + 256] = m1;
}

// ---------------------------------------------------------------------------
extern "C" void kernel_launch(void* const* d_in, const int* in_sizes, int n_in,
                              void* d_out, int out_size, void* d_ws, size_t ws_size,
                              hipStream_t stream) {
    const float* lat  = (const float*)d_in[0];   // [BATCH, HID]
    const float* W    = (const float*)d_in[1];   // [LATD, HID]
    const float* bias = (const float*)d_in[2];   // [LATD]
    // d_in[3] = n_sample (==1), ignored

    float* out_sampled = (float*)d_out;                          // [BATCH, LATD]
    float* out_mu      = out_sampled + (size_t)BATCH * LATD;     // [BATCH, LATD]
    float* out_norm    = out_mu + (size_t)BATCH * LATD;          // [BATCH]
    float* out_kld     = out_norm + BATCH;                       // [BATCH]

    // Stage 1: mu_raw -> out_mu (in-place raw; normalized by stage 2)
    dim3 g1(LATD / TILE_N, BATCH / TILE_M);
    gemm_bias_wmma<<<g1, 256, 0, stream>>>(lat, W, bias, out_mu);

    // Stage 2: per-row normalize + vMF sample
    vmf_epilogue<<<BATCH, 256, 0, stream>>>(out_mu, out_sampled, out_mu, out_norm, out_kld);
}
